// YyeAlignment_47863115546757
// MI455X (gfx1250) — matrix operand order repaired and checked by reading, not compiled, not run on hardware
//
#include <hip/hip_runtime.h>
#include <hip/hip_bf16.h>
#include <math.h>

// ---------------------------------------------------------------------------
// MI455X (gfx1250) pyramid/cascading deformable alignment.
// All conv3x3 / DCN-einsum GEMMs run on v_wmma_f32_16x16x32_f16 (fp32 acc).
// wave32; 8 waves/block; weights pre-packed to f16 WMMA fragments.
// ---------------------------------------------------------------------------

typedef __attribute__((ext_vector_type(16))) _Float16 v16h;
typedef __attribute__((ext_vector_type(8)))  _Float16 v8h;
typedef __attribute__((ext_vector_type(8)))  float    v8f;

#define NEG_SLOPE 0.1f

// ---------------------------------------------------------------------------
// Weight pre-pack: fp32 [M][K] -> f16 fragments, layout
//   Apk[ ((mt*KT + kt)*32 + lane)*16 + j ]
// so a lane's A fragment for (mt, kt) is one contiguous 32-byte v16h.
// A 16x32 fragment layout (lane half h, elem pair r):
//   K = (r<4 ? 2r : 16+2(r-4)) + 8h (+ j&1),  M = mt*16 + (lane&15)
// ---------------------------------------------------------------------------
__global__ void pack_w_kernel(const float* __restrict__ W,
                              _Float16* __restrict__ Apk,
                              int Mtot, int Ktot, int MTp)
{
    int KT  = Ktot >> 5;
    int tid = blockIdx.x * blockDim.x + threadIdx.x;
    int total = MTp * KT * 512;
    if (tid >= total) return;
    int j    = tid & 15;
    int lane = (tid >> 4) & 31;
    int kt   = (tid >> 9) % KT;
    int mt   = tid / (KT * 512);
    int half = lane >> 4, l16 = lane & 15;
    int r    = j >> 1;
    int klA  = (r < 4 ? 2 * r : 16 + 2 * (r - 4)) + half * 8 + (j & 1);
    int m    = mt * 16 + l16;
    int k    = kt * 32 + klA;
    float v  = (m < Mtot) ? W[(size_t)m * Ktot + k] : 0.f;
    Apk[tid] = (_Float16)v;
}

// ---------------------------------------------------------------------------
// Unified implicit-GEMM kernel.
//   Out[M][N] = act( W[M][K] * B[K][N] + bias ) (+res1)(+res2)
//   im2col=1 : B built on the fly, 3x3/pad1 over channel-concat (in0|in1).
//   im2col=0 : B = dense in0 ([K][N]) (DCN cols GEMM).
// Block 256 thr = 8 waves; tile 64(M) x 32(N); wave = one 16x16 WMMA tile.
// Requires N%32==0, K%32==0, W%4==0 (all true for this net).
// Producer: each thread owns k-row kk and 4 consecutive cols nn..nn+3;
// fully branchless (clamped indices + float masks, no exec divergence).
// ---------------------------------------------------------------------------
__global__ __launch_bounds__(256)
void conv_gemm_wmma(const _Float16* __restrict__ Apk,
                    const float* __restrict__ in0, int C0,
                    const float* __restrict__ in1, int C1,
                    const float* __restrict__ bias,
                    const float* __restrict__ res1,
                    const float* __restrict__ res2,
                    float* __restrict__ out,
                    int Mtot, int Ktot, int H, int Wd,
                    int im2col, int leaky)
{
    (void)C1;
    const int N  = H * Wd;
    const int KT = Ktot >> 5;
    const int colbase = blockIdx.x * 32;

    const int wave = threadIdx.x >> 5;
    const int lane = threadIdx.x & 31;
    const int miw  = wave >> 1;                 // 0..3 (M sub-tile)
    const int niw  = wave & 1;                  // 0..1 (N sub-tile)
    const int half = lane >> 4;
    const int l16  = lane & 15;
    const int mtile = blockIdx.y * 4 + miw;

    // double-buffered transposed B tile: [buf][n][k], 80B row stride
    __shared__ __align__(32) _Float16 ldsBT[2][32][40];

    // producer coordinates (loop-invariant)
    const int kk  = (threadIdx.x * 4) >> 5;     // k-row 0..31
    const int nn  = (threadIdx.x * 4) & 31;     // 0,4,...,28
    const int col = colbase + nn;               // multiple of 4; W%4==0 ->
    const int py  = col / Wd;                   // 4 cols share one image row
    const int px  = col - py * Wd;

    int c   = kk / 9;                           // k -> (channel, tap)
    int rem = kk - c * 9;

    v8f acc = {0.f, 0.f, 0.f, 0.f, 0.f, 0.f, 0.f, 0.f};

    const v16h* Ap = (const v16h*)Apk + (size_t)mtile * KT * 32 + lane;
    const int   nc = niw * 16 + l16;            // B fragment column

    for (int kt = 0; kt < KT; ++kt) {
        _Float16* bt = &ldsBT[kt & 1][0][0];
        // ---- produce 4 B values (k=kk, n=col..col+3), branchless ----
        float b0, b1, b2, b3;
        if (im2col) {
            int q  = (rem * 11) >> 5;           // rem / 3
            int ky = q - 1;
            int kx = rem - 3 * q - 1;
            int yy = py + ky;
            int xs = px + kx;
            const float* src = (c < C0) ? (in0 + (size_t)c * N)
                                        : (in1 + (size_t)(c - C0) * N);
            int   yc    = min(max(yy, 0), H - 1);
            float rowok = ((unsigned)yy < (unsigned)H) ? 1.f : 0.f;
            const float* rp = src + yc * Wd;
            int x0 = xs, x1 = xs + 1, x2 = xs + 2, x3 = xs + 3;
            b0 = rp[min(max(x0, 0), Wd - 1)] * (((unsigned)x0 < (unsigned)Wd) ? rowok : 0.f);
            b1 = rp[min(max(x1, 0), Wd - 1)] * (((unsigned)x1 < (unsigned)Wd) ? rowok : 0.f);
            b2 = rp[min(max(x2, 0), Wd - 1)] * (((unsigned)x2 < (unsigned)Wd) ? rowok : 0.f);
            b3 = rp[min(max(x3, 0), Wd - 1)] * (((unsigned)x3 < (unsigned)Wd) ? rowok : 0.f);
            // advance k by 32 = 3*9 + 5 (branchless)
            int wrap = (rem + 5 >= 9) ? 1 : 0;
            rem = rem + 5 - 9 * wrap;
            c  += 3 + wrap;
        } else {
            const float4 bv = *(const float4*)(in0 + (size_t)(kt * 32 + kk) * N + col);
            b0 = bv.x; b1 = bv.y; b2 = bv.z; b3 = bv.w;
        }
        // transposed store: bt[n*40 + k]
        bt[(nn + 0) * 40 + kk] = (_Float16)b0;
        bt[(nn + 1) * 40 + kk] = (_Float16)b1;
        bt[(nn + 2) * 40 + kk] = (_Float16)b2;
        bt[(nn + 3) * 40 + kk] = (_Float16)b3;
        __syncthreads();

        // ---- fragments: A = one 32B packed load; B = two 16B LDS loads ----
        v16h afrag = Ap[(size_t)kt * 32];
        const _Float16* brow = bt + nc * 40 + 8 * half;
        v8h blo = *(const v8h*)(brow);
        v8h bhi = *(const v8h*)(brow + 16);
        v16h bfrag = __builtin_shufflevector(blo, bhi,
                        0, 1, 2, 3, 4, 5, 6, 7, 8, 9, 10, 11, 12, 13, 14, 15);

        acc = __builtin_amdgcn_wmma_f32_16x16x32_f16(
                  false, afrag, false, bfrag, (short)0, acc, false, false);
        // no second barrier: double-buffered LDS
    }

    // ---- epilogue: bias -> leaky -> residual adds ----
    #pragma unroll
    for (int r = 0; r < 8; ++r) {
        int m    = mtile * 16 + half * 8 + r;
        int colw = colbase + niw * 16 + l16;
        if (m < Mtot) {
            float v = acc[r];
            if (bias)  v += bias[m];
            if (leaky) v = (v > 0.f) ? v : NEG_SLOPE * v;
            size_t o = (size_t)m * N + colw;
            if (res1) v += res1[o];
            if (res2) v += res2[o];
            out[o] = v;
        }
    }
}

// ---------------------------------------------------------------------------
// Deformable bilinear gather: cols[(g*8+c)*9+k][hw] = bilin(x)(py,px) * mask.
// One thread per (g,k,hw), loops over Cg=8 channels sharing coords/mask.
// ---------------------------------------------------------------------------
__global__ void build_cols_kernel(const float* __restrict__ xin,  // [64][HW]
                                  const float* __restrict__ off,  // [216][HW]
                                  float* __restrict__ cols,       // [576][HW]
                                  int H, int Wd)
{
    const int N = H * Wd;
    int tid = blockIdx.x * blockDim.x + threadIdx.x;
    int total = 8 * 9 * N;
    if (tid >= total) return;

    int hw = tid % N;
    int gk = tid / N;
    int g  = gk / 9;
    int k  = gk - g * 9;
    int y  = hw / Wd;
    int x  = hw - y * Wd;

    float oy = off[(size_t)gk * N + hw];
    float ox = off[(size_t)(72 + gk) * N + hw];
    float mv = off[(size_t)(144 + gk) * N + hw];
    float msk = 1.f / (1.f + expf(-mv));

    float py = (float)y + (float)(k / 3 - 1) + oy;
    float px = (float)x + (float)(k % 3 - 1) + ox;
    float y0f = floorf(py), x0f = floorf(px);
    float dy = py - y0f, dx = px - x0f;
    int y0 = (int)y0f, x0 = (int)x0f;
    int y1 = y0 + 1,   x1 = x0 + 1;

    float vy0 = (y0 >= 0 && y0 < H) ? 1.f : 0.f;
    float vy1 = (y1 >= 0 && y1 < H) ? 1.f : 0.f;
    float vx0 = (x0 >= 0 && x0 < Wd) ? 1.f : 0.f;
    float vx1 = (x1 >= 0 && x1 < Wd) ? 1.f : 0.f;

    int y0c = min(max(y0, 0), H - 1),  y1c = min(max(y1, 0), H - 1);
    int x0c = min(max(x0, 0), Wd - 1), x1c = min(max(x1, 0), Wd - 1);

    float w00 = (1.f - dy) * (1.f - dx) * vy0 * vx0;
    float w01 = (1.f - dy) * dx        * vy0 * vx1;
    float w10 = dy        * (1.f - dx) * vy1 * vx0;
    float w11 = dy        * dx         * vy1 * vx1;

    int i00 = y0c * Wd + x0c, i01 = y0c * Wd + x1c;
    int i10 = y1c * Wd + x0c, i11 = y1c * Wd + x1c;

    #pragma unroll
    for (int cch = 0; cch < 8; ++cch) {
        const float* src = xin + (size_t)(g * 8 + cch) * N;
        float v = w00 * src[i00] + w01 * src[i01]
                + w10 * src[i10] + w11 * src[i11];
        cols[(size_t)((g * 8 + cch) * 9 + k) * N + hw] = v * msk;
    }
}

// Bilinear 2x upsample (half-pixel centers, clamp-to-edge) times 2.0
__global__ void up2_bilinear_x2(const float* __restrict__ in,
                                float* __restrict__ out,
                                int C, int Hin, int Win)
{
    int Hout = Hin * 2, Wout = Win * 2;
    int tot = C * Hout * Wout;
    int tid = blockIdx.x * blockDim.x + threadIdx.x;
    if (tid >= tot) return;
    int x = tid % Wout;
    int t = tid / Wout;
    int y = t % Hout;
    int c = t / Hout;

    float sy = (y + 0.5f) * 0.5f - 0.5f;
    float sx = (x + 0.5f) * 0.5f - 0.5f;
    int y0 = (int)floorf(sy);
    int x0 = (int)floorf(sx);
    float dy = sy - (float)y0;
    float dx = sx - (float)x0;
    int y0c = min(max(y0, 0), Hin - 1), y1c = min(max(y0 + 1, 0), Hin - 1);
    int x0c = min(max(x0, 0), Win - 1), x1c = min(max(x0 + 1, 0), Win - 1);
    const float* p = in + (size_t)c * Hin * Win;
    float v = (1.f - dy) * ((1.f - dx) * p[y0c * Win + x0c] + dx * p[y0c * Win + x1c])
            + dy        * ((1.f - dx) * p[y1c * Win + x0c] + dx * p[y1c * Win + x1c]);
    out[tid] = 2.f * v;
}

// 2x2 mean pool over CH contiguous channel maps
__global__ void pool2_kernel(const float* __restrict__ in,
                             float* __restrict__ out,
                             int CH, int Hin, int Win)
{
    int Hout = Hin / 2, Wout = Win / 2;
    int tot = CH * Hout * Wout;
    int tid = blockIdx.x * blockDim.x + threadIdx.x;
    if (tid >= tot) return;
    int x = tid % Wout;
    int t = tid / Wout;
    int y = t % Hout;
    int c = t / Hout;
    const float* p = in + (size_t)c * Hin * Win + (size_t)(2 * y) * Win + 2 * x;
    out[tid] = 0.25f * (p[0] + p[1] + p[Win] + p[Win + 1]);
}

__global__ void copy_kernel(const float* __restrict__ in,
                            float* __restrict__ out, int n)
{
    int tid = blockIdx.x * blockDim.x + threadIdx.x;
    if (tid < n) out[tid] = in[tid];
}

// ---------------------------------------------------------------------------
// Host orchestration
// ---------------------------------------------------------------------------
static inline void launch_conv(hipStream_t s, const _Float16* Apk,
                               const float* in0, int C0,
                               const float* in1, int C1,
                               const float* bias,
                               const float* res1, const float* res2,
                               float* out, int M, int K, int H, int Wd,
                               int im2col, int leaky)
{
    int N = H * Wd;
    dim3 grid((unsigned)(N / 32), (unsigned)((M + 63) / 64));
    conv_gemm_wmma<<<grid, 256, 0, s>>>(Apk, in0, C0, in1, C1, bias, res1, res2,
                                        out, M, K, H, Wd, im2col, leaky);
}

extern "C" void kernel_launch(void* const* d_in, const int* in_sizes, int n_in,
                              void* d_out, int out_size, void* d_ws, size_t ws_size,
                              hipStream_t stream)
{
    (void)in_sizes; (void)n_in; (void)out_size; (void)ws_size;

    const int T = 5, C = 64, H = 96, W = 96, tc = 2;
    const int HW = H * W;                         // 9216
    const int H2 = 48, W2 = 48, HW2 = H2 * W2;    // 2304
    const int H3 = 24, W3 = 24, HW3 = H3 * W3;    // 576

    // -------- inputs (setup_inputs dict order, nested dicts in order) ------
    int ip = 0;
    auto nxt = [&]() { return (const float*)d_in[ip++]; };
    const float* X    = nxt();
    const float *a3w = nxt(), *a3b = nxt();
    const float *b3w = nxt(), *b3b = nxt();
    const float *a2w = nxt(), *a2b = nxt();
    const float *b2w = nxt(), *b2b = nxt();
    const float *c2w = nxt(), *c2bi = nxt();
    const float *d2wo = nxt(), *d2bo = nxt(), *d2w = nxt(), *d2b = nxt();
    const float *a1w = nxt(), *a1b = nxt();
    const float *b1w = nxt(), *b1b = nxt();
    const float *c1w = nxt(), *c1bi = nxt();
    const float *d1wo = nxt(), *d1bo = nxt(), *d1w = nxt(), *d1b = nxt();
    const float *dfwo = nxt(), *dfbo = nxt(), *dfw = nxt(), *dfb = nxt();
    const float *caw = nxt(), *cab = nxt();
    const float *cbw = nxt(), *cbb = nxt();
    const float *cdwo = nxt(), *cdbo = nxt(), *cdw = nxt(), *cdb = nxt();

    float* out = (float*)d_out;

    // -------- workspace layout (floats, 32B-aligned slices) ----------------
    float* ws = (float*)d_ws;
    size_t off = 0;
    auto alloc = [&](size_t n) {
        off = (off + 7) & ~(size_t)7;
        float* p = ws + off; off += n; return p;
    };
    float* x2buf = alloc((size_t)T * C * HW2);
    float* x3buf = alloc((size_t)T * C * HW3);
    float* offcv = alloc((size_t)216 * HW);       // DCN offset conv (shared)
    float* cols  = alloc((size_t)576 * HW);       // DCN cols (shared)
    float* t3a   = alloc((size_t)C * HW3);
    float* t3b   = alloc((size_t)C * HW3);
    float* u3    = alloc((size_t)C * HW2);
    float* fa2   = alloc((size_t)C * HW2);
    float* bc2   = alloc((size_t)C * HW2);
    float* t2    = alloc((size_t)C * HW2);
    float* o2    = alloc((size_t)C * HW2);
    float* u2    = alloc((size_t)C * HW);
    float* fa1   = alloc((size_t)C * HW);
    float* bc1   = alloc((size_t)C * HW);
    float* t1    = alloc((size_t)C * HW);
    float* o1    = alloc((size_t)C * HW);
    float* faf   = alloc((size_t)C * HW);
    float* offa  = alloc((size_t)C * HW);
    float* offb  = alloc((size_t)C * HW);

    // -------- pack all layer weights to f16 WMMA fragments -----------------
    auto pack = [&](const float* Wsrc, int M, int K) -> const _Float16* {
        int MTp = ((M + 63) / 64) * 4;
        int KT  = K / 32;
        size_t elems  = (size_t)MTp * KT * 512;       // halves
        float* buf = alloc((elems + 1) / 2);
        _Float16* dst = (_Float16*)buf;
        pack_w_kernel<<<(unsigned)((elems + 255) / 256), 256, 0, stream>>>(
            Wsrc, dst, M, K, MTp);
        return dst;
    };
    const _Float16 *Pa3  = pack(a3w, 64, 1152), *Pb3 = pack(b3w, 64, 576);
    const _Float16 *Pa2  = pack(a2w, 64, 1152), *Pb2 = pack(b2w, 64, 1152);
    const _Float16 *Pc2  = pack(c2w, 64, 576);
    const _Float16 *Pd2o = pack(d2wo, 216, 576), *Pd2 = pack(d2w, 64, 576);
    const _Float16 *Pa1  = pack(a1w, 64, 1152), *Pb1 = pack(b1w, 64, 1152);
    const _Float16 *Pc1  = pack(c1w, 64, 576);
    const _Float16 *Pd1o = pack(d1wo, 216, 576), *Pd1 = pack(d1w, 64, 576);
    const _Float16 *Pdfo = pack(dfwo, 216, 576), *Pdf = pack(dfw, 64, 576);
    const _Float16 *Pca  = pack(caw, 64, 1152), *Pcb = pack(cbw, 64, 576);
    const _Float16 *Pcdo = pack(cdwo, 216, 576), *Pcd = pack(cdw, 64, 576);

    // -------- pyramid ------------------------------------------------------
    {
        int tot2 = T * C * HW2;
        pool2_kernel<<<(tot2 + 255) / 256, 256, 0, stream>>>(X, x2buf, T * C, H, W);
        int tot3 = T * C * HW3;
        pool2_kernel<<<(tot3 + 255) / 256, 256, 0, stream>>>(x2buf, x3buf, T * C, H2, W2);
    }

    const float* f1c = X     + (size_t)tc * C * HW;
    const float* f2c = x2buf + (size_t)tc * C * HW2;
    const float* f3c = x3buf + (size_t)tc * C * HW3;

    // center frame passthrough
    {
        int n = C * HW;
        copy_kernel<<<(n + 255) / 256, 256, 0, stream>>>(
            f1c, out + (size_t)tc * C * HW, n);
    }

    for (int i = 0; i < T; ++i) {
        if (i == tc) continue;
        const float* x1i = X     + (size_t)i * C * HW;
        const float* x2i = x2buf + (size_t)i * C * HW2;
        const float* x3i = x3buf + (size_t)i * C * HW3;

        // ---- level 3: off3 = cm(cm(cat(x3_i, f3c)))
        launch_conv(stream, Pa3, x3i, C, f3c, C, a3b, nullptr, nullptr,
                    t3a, 64, 1152, H3, W3, 1, 1);
        launch_conv(stream, Pb3, t3a, C, nullptr, 0, b3b, nullptr, nullptr,
                    t3b, 64, 576, H3, W3, 1, 1);
        {
            int tot = C * HW2;
            up2_bilinear_x2<<<(tot + 255) / 256, 256, 0, stream>>>(t3b, u3, C, H3, W3);
        }

        // ---- level 2: fa2 = mdcn(x2_i, u3, dcn2)
        launch_conv(stream, Pd2o, u3, C, nullptr, 0, d2bo, nullptr, nullptr,
                    offcv, 216, 576, H2, W2, 1, 0);
        {
            int tot = 8 * 9 * HW2;
            build_cols_kernel<<<(tot + 255) / 256, 256, 0, stream>>>(
                x2i, offcv, cols, H2, W2);
        }
        launch_conv(stream, Pd2, cols, 0, nullptr, 0, d2b, nullptr, nullptr,
                    fa2, 64, 576, H2, W2, 0, 0);
        launch_conv(stream, Pa2, fa2, C, f2c, C, a2b, nullptr, nullptr,
                    bc2, 64, 1152, H2, W2, 1, 1);
        launch_conv(stream, Pb2, bc2, C, u3, C, b2b, nullptr, nullptr,
                    t2, 64, 1152, H2, W2, 1, 1);
        launch_conv(stream, Pc2, t2, C, nullptr, 0, c2bi, u3, bc2,
                    o2, 64, 576, H2, W2, 1, 1);
        {
            int tot = C * HW;
            up2_bilinear_x2<<<(tot + 255) / 256, 256, 0, stream>>>(o2, u2, C, H2, W2);
        }

        // ---- level 1: fa1 = mdcn(x1_i, u2, dcn1)
        launch_conv(stream, Pd1o, u2, C, nullptr, 0, d1bo, nullptr, nullptr,
                    offcv, 216, 576, H, W, 1, 0);
        {
            int tot = 8 * 9 * HW;
            build_cols_kernel<<<(tot + 255) / 256, 256, 0, stream>>>(
                x1i, offcv, cols, H, W);
        }
        launch_conv(stream, Pd1, cols, 0, nullptr, 0, d1b, nullptr, nullptr,
                    fa1, 64, 576, H, W, 0, 0);
        launch_conv(stream, Pa1, fa1, C, f1c, C, a1b, nullptr, nullptr,
                    bc1, 64, 1152, H, W, 1, 1);
        launch_conv(stream, Pb1, bc1, C, u2, C, b1b, nullptr, nullptr,
                    t1, 64, 1152, H, W, 1, 1);
        launch_conv(stream, Pc1, t1, C, nullptr, 0, c1bi, u2, bc1,
                    o1, 64, 576, H, W, 1, 1);

        // ---- final cascade: faf = mdcn(x1_i, o1, dcnf)
        launch_conv(stream, Pdfo, o1, C, nullptr, 0, dfbo, nullptr, nullptr,
                    offcv, 216, 576, H, W, 1, 0);
        {
            int tot = 8 * 9 * HW;
            build_cols_kernel<<<(tot + 255) / 256, 256, 0, stream>>>(
                x1i, offcv, cols, H, W);
        }
        launch_conv(stream, Pdf, cols, 0, nullptr, 0, dfb, nullptr, nullptr,
                    faf, 64, 576, H, W, 0, 0);
        launch_conv(stream, Pca, faf, C, f1c, C, cab, nullptr, nullptr,
                    offa, 64, 1152, H, W, 1, 1);
        launch_conv(stream, Pcb, offa, C, nullptr, 0, cbb, nullptr, nullptr,
                    offb, 64, 576, H, W, 1, 1);
        launch_conv(stream, Pcdo, offb, C, nullptr, 0, cdbo, nullptr, nullptr,
                    offcv, 216, 576, H, W, 1, 0);
        {
            int tot = 8 * 9 * HW;
            build_cols_kernel<<<(tot + 255) / 256, 256, 0, stream>>>(
                faf, offcv, cols, H, W);
        }
        launch_conv(stream, Pcd, cols, 0, nullptr, 0, cdb, nullptr, nullptr,
                    out + (size_t)i * C * HW, 64, 576, H, W, 0, 1);
    }
}
